// StateSpace2dJoints_33363305955827
// MI455X (gfx1250) — matrix operand order repaired and checked by reading, not compiled, not run on hardware
//
#include <hip/hip_runtime.h>

typedef __attribute__((ext_vector_type(16))) _Float16 v16h;
typedef __attribute__((ext_vector_type(8)))  _Float16 v8h;
typedef __attribute__((ext_vector_type(8)))  float    v8f;

#define IN_DIM   34
#define ENC_DIM  128
#define RNN_DIM  64
#define GATE_DIM 256
#define N_SEEDS  10
#define T_TOT    60
#define N_DEC    50
#define BATCH    8192

#define WAVES_PER_BLOCK 4
#define ROWS_PER_WAVE   16
#define ROWS_PER_BLOCK  (WAVES_PER_BLOCK * ROWS_PER_WAVE)

__device__ __forceinline__ v8f wmma16(v16h a, v16h b, v8f c) {
  // 8 args: (neg_a, A, neg_b, B, c_mod, C, reuse_a, reuse_b)
  return __builtin_amdgcn_wmma_f32_16x16x32_f16(false, a, false, b, (short)0, c,
                                                false, false);
}

// Wave-local LDS producer->consumer fence: all buffers touched inside the time
// loop are private to this wave; same-wave LDS ops complete in order, so
// s_wait_dscnt 0 is sufficient (no block barrier needed).
__device__ __forceinline__ void lds_fence_wave() {
  asm volatile("s_wait_dscnt 0" ::: "memory");
}

// A fragment (16x32, f16) from a row-major f16 LDS buffer.
// ISA layout: lane m (0-15) / m (16-31 -> g=1); element e -> K = (e&7) + 8*g + 16*(e>>3)
// => two contiguous 8-half (16B) loads at k0+8g and k0+16+8g.
__device__ __forceinline__ v16h load_a_frag(const _Float16* buf, int strideH,
                                            int k0, int lane) {
  int m = lane & 15, g = lane >> 4;
  const _Float16* p = buf + m * strideH + k0 + 8 * g;
  v8h lo = *(const v8h*)p;
  v8h hi = *(const v8h*)(p + 16);
  v16h r;
#pragma unroll
  for (int i = 0; i < 8; ++i) { r[i] = lo[i]; r[i + 8] = hi[i]; }
  return r;
}

// B fragment (32x16, f16) from an n-major f16 LDS weight image W[n][k].
// ISA layout: lane = column n (g = lane>>4); element e -> K = (e&7) + 8*g + 16*(e>>3)
// => two contiguous 8-half (16B) loads at k0+8g and k0+16+8g.
__device__ __forceinline__ v16h load_b_frag(const _Float16* buf, int strideH,
                                            int n0, int k0, int lane) {
  int n = lane & 15, g = lane >> 4;
  const _Float16* p = buf + (n0 + n) * strideH + k0 + 8 * g;
  v8h lo = *(const v8h*)p;
  v8h hi = *(const v8h*)(p + 16);
  v16h r;
#pragma unroll
  for (int i = 0; i < 8; ++i) { r[i] = lo[i]; r[i + 8] = hi[i]; }
  return r;
}

__device__ __forceinline__ float fast_sigmoid(float x) {
  return __builtin_amdgcn_rcpf(1.0f + __expf(-x));
}
__device__ __forceinline__ float fast_tanh(float x) {
  x = fminf(15.0f, fmaxf(-15.0f, x));
  float e = __expf(2.0f * x);
  return (e - 1.0f) * __builtin_amdgcn_rcpf(e + 1.0f);
}

__global__ __launch_bounds__(ROWS_PER_BLOCK * 2, 1)
void lstm_seq_wmma_kernel(const float* __restrict__ x,
                          const float* __restrict__ enc_W,
                          const float* __restrict__ enc_b,
                          const float* __restrict__ W_ih,
                          const float* __restrict__ W_hh,
                          const float* __restrict__ b_ih,
                          const float* __restrict__ b_hh,
                          const float* __restrict__ dec_W,
                          const float* __restrict__ dec_b,
                          float* __restrict__ out) {
  // f16 weight images, n-major, K padded where needed
  __shared__ __align__(16) _Float16 sWih[GATE_DIM * ENC_DIM];   // 64 KB
  __shared__ __align__(16) _Float16 sWhh[GATE_DIM * RNN_DIM];   // 32 KB
  __shared__ __align__(16) _Float16 sEnc[ENC_DIM * 64];         // 16 KB (K 34->64)
  __shared__ __align__(16) _Float16 sDec[48 * RNN_DIM];         //  6 KB (N 34->48)
  __shared__ __align__(16) _Float16 sXpad[WAVES_PER_BLOCK][16 * 64];       // 8 KB
  __shared__ __align__(16) _Float16 sEmb [WAVES_PER_BLOCK][16 * ENC_DIM];  // 16 KB
  __shared__ __align__(16) _Float16 sH   [WAVES_PER_BLOCK][16 * RNN_DIM];  // 8 KB

  const int tid = threadIdx.x;

  // ---- stage weights (f32 -> f16) into LDS ----
  for (int i = tid; i < GATE_DIM * ENC_DIM; i += blockDim.x)
    sWih[i] = (_Float16)W_ih[i];                       // already [n][k]
  for (int i = tid; i < GATE_DIM * RNN_DIM; i += blockDim.x)
    sWhh[i] = (_Float16)W_hh[i];
  for (int i = tid; i < ENC_DIM * 64; i += blockDim.x) {
    int nn = i >> 6, kk = i & 63;
    sEnc[i] = (kk < IN_DIM) ? (_Float16)enc_W[nn * IN_DIM + kk] : (_Float16)0.0f;
  }
  for (int i = tid; i < 48 * RNN_DIM; i += blockDim.x) {
    int nn = i >> 6, kk = i & 63;
    sDec[i] = (nn < IN_DIM) ? (_Float16)dec_W[nn * RNN_DIM + kk] : (_Float16)0.0f;
  }
  __syncthreads();  // only block-wide barrier: weights are read-only afterwards

  const int wave = tid >> 5;
  const int lane = tid & 31;
  const int n    = lane & 15;
  const int g    = lane >> 4;
  const int row0 = (blockIdx.x * WAVES_PER_BLOCK + wave) * ROWS_PER_WAVE;

  _Float16* xpad = sXpad[wave];
  _Float16* emb  = sEmb[wave];
  _Float16* hbuf = sH[wave];

  // ---- per-lane bias registers (bias depends only on column n) ----
  float encb[8], gb[16], db[3];
#pragma unroll
  for (int tset = 0; tset < 8; ++tset) encb[tset] = enc_b[tset * 16 + n];
#pragma unroll
  for (int tset = 0; tset < 16; ++tset)
    gb[tset] = b_ih[tset * 16 + n] + b_hh[tset * 16 + n];
#pragma unroll
  for (int tset = 0; tset < 3; ++tset) {
    int col = tset * 16 + n;
    db[tset] = (col < IN_DIM) ? dec_b[col] : 0.0f;
  }

  // ---- hoist loop-invariant decoder B fragments (6 x 8 VGPRs) ----
  v16h dB0[3], dB1[3];
#pragma unroll
  for (int nt = 0; nt < 3; ++nt) {
    dB0[nt] = load_b_frag(sDec, RNN_DIM, nt * 16, 0, lane);
    dB1[nt] = load_b_frag(sDec, RNN_DIM, nt * 16, 32, lane);
  }

  // ---- init state: h = c = 0; prev = x[:, N_SEEDS-1, :] ----
  for (int i = lane; i < 16 * RNN_DIM; i += 32) hbuf[i] = (_Float16)0.0f;
  v8f cst[4];
#pragma unroll
  for (int jt = 0; jt < 4; ++jt)
#pragma unroll
    for (int r = 0; r < 8; ++r) cst[jt][r] = 0.0f;

  v8f prev[3];
#pragma unroll
  for (int nt = 0; nt < 3; ++nt) {
    int col = nt * 16 + n;
#pragma unroll
    for (int r = 0; r < 8; ++r) {
      int m = r + 8 * g;
      prev[nt][r] = (col < IN_DIM)
          ? x[(size_t)(row0 + m) * (T_TOT * IN_DIM) + (N_SEEDS - 1) * IN_DIM + col]
          : 0.0f;
    }
  }
  lds_fence_wave();  // hbuf zeros + dB fragments resolved

  // ---- time loop (waves run fully independently from here on) ----
  for (int t = 0; t < T_TOT; ++t) {
    const bool seed = (t < N_SEEDS);

    v16h eA[4];
    if (seed) {
      // stage x[:, t, :] padded to K=64 as f16
      for (int idx = lane; idx < 16 * 64; idx += 32) {
        int r = idx >> 6, cc = idx & 63;
        float v = (cc < IN_DIM)
            ? x[(size_t)(row0 + r) * (T_TOT * IN_DIM) + t * IN_DIM + cc]
            : 0.0f;
        xpad[idx] = (_Float16)v;
      }
      lds_fence_wave();

      v16h xA0 = load_a_frag(xpad, 64, 0, lane);
      v16h xA1 = load_a_frag(xpad, 64, 32, lane);
      // encoder: emb = relu(x @ enc_W^T + enc_b), 8 N-tiles of 16
#pragma unroll
      for (int nt = 0; nt < 8; ++nt) {
        v8f acc;
#pragma unroll
        for (int r = 0; r < 8; ++r) acc[r] = encb[nt];
        acc = wmma16(xA0, load_b_frag(sEnc, 64, nt * 16, 0, lane), acc);
        acc = wmma16(xA1, load_b_frag(sEnc, 64, nt * 16, 32, lane), acc);
#pragma unroll
        for (int r = 0; r < 8; ++r) {
          int m = r + 8 * g;
          emb[m * ENC_DIM + nt * 16 + n] = (_Float16)fmaxf(acc[r], 0.0f);
        }
      }
      lds_fence_wave();
#pragma unroll
      for (int kk = 0; kk < 4; ++kk) eA[kk] = load_a_frag(emb, ENC_DIM, kk * 32, lane);
    }

    // old h as A fragments (registers) before hbuf is overwritten
    v16h hA0 = load_a_frag(hbuf, RNN_DIM, 0, lane);
    v16h hA1 = load_a_frag(hbuf, RNN_DIM, 32, lane);

    // gates in 4 column groups of 64 (tiles jt, jt+4, jt+8, jt+12 = i,f,g,o)
#pragma unroll
    for (int jt = 0; jt < 4; ++jt) {
      v8f gi, gf, gg_, go;
#pragma unroll
      for (int r = 0; r < 8; ++r) {
        gi[r]  = gb[jt];
        gf[r]  = gb[jt + 4];
        gg_[r] = gb[jt + 8];
        go[r]  = gb[jt + 12];
      }
      if (seed) {
#pragma unroll
        for (int kk = 0; kk < 4; ++kk) {
          gi  = wmma16(eA[kk], load_b_frag(sWih, ENC_DIM, (jt)*16,      kk * 32, lane), gi);
          gf  = wmma16(eA[kk], load_b_frag(sWih, ENC_DIM, (jt + 4)*16,  kk * 32, lane), gf);
          gg_ = wmma16(eA[kk], load_b_frag(sWih, ENC_DIM, (jt + 8)*16,  kk * 32, lane), gg_);
          go  = wmma16(eA[kk], load_b_frag(sWih, ENC_DIM, (jt + 12)*16, kk * 32, lane), go);
        }
      }
      gi  = wmma16(hA0, load_b_frag(sWhh, RNN_DIM, (jt)*16,      0, lane), gi);
      gf  = wmma16(hA0, load_b_frag(sWhh, RNN_DIM, (jt + 4)*16,  0, lane), gf);
      gg_ = wmma16(hA0, load_b_frag(sWhh, RNN_DIM, (jt + 8)*16,  0, lane), gg_);
      go  = wmma16(hA0, load_b_frag(sWhh, RNN_DIM, (jt + 12)*16, 0, lane), go);
      gi  = wmma16(hA1, load_b_frag(sWhh, RNN_DIM, (jt)*16,      32, lane), gi);
      gf  = wmma16(hA1, load_b_frag(sWhh, RNN_DIM, (jt + 4)*16,  32, lane), gf);
      gg_ = wmma16(hA1, load_b_frag(sWhh, RNN_DIM, (jt + 8)*16,  32, lane), gg_);
      go  = wmma16(hA1, load_b_frag(sWhh, RNN_DIM, (jt + 12)*16, 32, lane), go);

      // LSTM cell elementwise (PyTorch gate order i,f,g,o)
#pragma unroll
      for (int r = 0; r < 8; ++r) {
        float ii  = fast_sigmoid(gi[r]);
        float ff  = fast_sigmoid(gf[r]);
        float ggv = fast_tanh(gg_[r]);
        float oo  = fast_sigmoid(go[r]);
        float cv  = ff * cst[jt][r] + ii * ggv;
        cst[jt][r] = cv;
        float hv = oo * fast_tanh(cv);
        hbuf[(r + 8 * g) * RNN_DIM + jt * 16 + n] = (_Float16)hv;
      }
    }
    lds_fence_wave();  // new h committed for next step / decoder projection

    if (!seed) {
      v16h h2A0 = load_a_frag(hbuf, RNN_DIM, 0, lane);
      v16h h2A1 = load_a_frag(hbuf, RNN_DIM, 32, lane);
#pragma unroll
      for (int nt = 0; nt < 3; ++nt) {
        v8f y;
#pragma unroll
        for (int r = 0; r < 8; ++r) y[r] = prev[nt][r] + db[nt];
        y = wmma16(h2A0, dB0[nt], y);
        y = wmma16(h2A1, dB1[nt], y);
        prev[nt] = y;
        int col = nt * 16 + n;
        if (col < IN_DIM) {
#pragma unroll
          for (int r = 0; r < 8; ++r) {
            int m = r + 8 * g;
            out[(size_t)(row0 + m) * (N_DEC * IN_DIM) + (t - N_SEEDS) * IN_DIM + col] = y[r];
          }
        }
      }
    }
  }
}

extern "C" void kernel_launch(void* const* d_in, const int* in_sizes, int n_in,
                              void* d_out, int out_size, void* d_ws, size_t ws_size,
                              hipStream_t stream) {
  const float* x     = (const float*)d_in[0];
  const float* enc_W = (const float*)d_in[1];
  const float* enc_b = (const float*)d_in[2];
  const float* W_ih  = (const float*)d_in[3];
  const float* W_hh  = (const float*)d_in[4];
  const float* b_ih  = (const float*)d_in[5];
  const float* b_hh  = (const float*)d_in[6];
  const float* dec_W = (const float*)d_in[7];
  const float* dec_b = (const float*)d_in[8];
  float* out = (float*)d_out;

  dim3 grid(BATCH / ROWS_PER_BLOCK);           // 128 blocks
  dim3 block(WAVES_PER_BLOCK * 32);            // 4 waves (wave32)
  hipLaunchKernelGGL(lstm_seq_wmma_kernel, grid, block, 0, stream,
                     x, enc_W, enc_b, W_ih, W_hh, b_ih, b_hh, dec_W, dec_b, out);
}
